// TissueGraphEncoder_21483426414738
// MI455X (gfx1250) — compile-verified, matrix operand order checked by reading, not run on hardware
//
#include <hip/hip_runtime.h>
#include <hip/hip_bf16.h>

#define DIM 128
#define LN_EPS 1e-5f
#define EPW 4  // edges per wave in scatter kernel

typedef float    v2f  __attribute__((ext_vector_type(2)));
typedef float    v8f  __attribute__((ext_vector_type(8)));
typedef _Float16 v16h __attribute__((ext_vector_type(16)));

#if __has_builtin(__builtin_amdgcn_wmma_f32_16x16x4_f32)
#define USE_F32_WMMA 1
#else
#define USE_F32_WMMA 0
#endif

// ---------------- degree / normalization ----------------

__global__ __launch_bounds__(256) void k_deg_init(float* __restrict__ deg, int n) {
  int i = blockIdx.x * 256 + threadIdx.x;
  if (i < n) deg[i] = 1.0f;  // self-loop
}

__global__ __launch_bounds__(256) void k_deg_edges(const int* __restrict__ dst,
                                                   float* __restrict__ deg, int e) {
  int i = blockIdx.x * 256 + threadIdx.x;
  if (i < e) atomicAdd(&deg[dst[i]], 1.0f);
}

__global__ __launch_bounds__(256) void k_deg_rsqrt(float* __restrict__ deg, int n) {
  int i = blockIdx.x * 256 + threadIdx.x;
  if (i < n) deg[i] = rsqrtf(deg[i]);  // deg buffer now holds dinv
}

// ---------------- GEMM: H = X * W via WMMA ----------------
// One wave computes a 16-row x 128-col tile. W is staged TRANSPOSED in LDS with an
// XOR bank swizzle: element (n,k) lives at sWT[n*128 + (k ^ ((n&15)<<2))].
//  - B fragment per (kt,nt) = one aligned ds_load_b64 per lane (v2f), no reg shuffles
//  - swizzle keeps 8B pair contiguity (XOR bits >= 2) and spreads the 32-lane
//    fragment read across all 64 LDS banks exactly once.

__global__ __launch_bounds__(256) void k_gemm_wmma(const float* __restrict__ X,
                                                   const float* __restrict__ W,
                                                   float* __restrict__ H,
                                                   int nTiles) {
  __shared__ float sWT[DIM * DIM];  // 64 KB
  {
    // W is [k][n] row-major; write transposed+swizzled.
    for (int t = threadIdx.x; t < DIM * DIM; t += 256) {
      int k = t >> 7;        // row of W
      int n = t & 127;       // col of W (consecutive across threads -> coalesced read)
      sWT[n * DIM + (k ^ ((n & 15) << 2))] = W[t];
    }
  }
  __syncthreads();

  const int wave = threadIdx.x >> 5;
  const int lane = threadIdx.x & 31;
  const int tile = blockIdx.x * 8 + wave;
  if (tile >= nTiles) return;

  const int m  = lane & 15;   // row within 16-row tile (and col within 16-col tile)
  const int kh = lane >> 4;   // K half selector
  const int kxor = m << 2;    // per-lane constant swizzle
  const float* __restrict__ xrow = X + (size_t)(tile * 16 + m) * DIM;

  v8f acc[8];
  const v8f vzero = {0.f, 0.f, 0.f, 0.f, 0.f, 0.f, 0.f, 0.f};
#pragma unroll
  for (int nt = 0; nt < 8; ++nt) acc[nt] = vzero;

#if USE_F32_WMMA
  // fp32 WMMA: 16x16x4, K=128 -> 32 steps, full fp32 precision.
  for (int kt = 0; kt < DIM / 4; ++kt) {
    const int k0 = kt * 4 + kh * 2;
    v2f a = *(const v2f*)&xrow[k0];
    const int kx = k0 ^ kxor;  // swizzled k (pair-contiguous, 8B aligned)
#pragma unroll
    for (int nt = 0; nt < 8; ++nt) {
      const int n = nt * 16 + m;
      v2f b = *(const v2f*)&sWT[n * DIM + kx];
      acc[nt] = __builtin_amdgcn_wmma_f32_16x16x4_f32(
          false, a, false, b, (short)0, acc[nt], false, false);
    }
  }
#else
  // fallback: f16 inputs, f32 accumulate, 16x16x32
  for (int kt = 0; kt < DIM / 32; ++kt) {
    v16h a;
#pragma unroll
    for (int e = 0; e < 16; ++e) {
      const int k = kt * 32 + ((e < 8) ? (kh * 8 + e) : (16 + kh * 8 + (e - 8)));
      a[e] = (_Float16)xrow[k];
    }
#pragma unroll
    for (int nt = 0; nt < 8; ++nt) {
      const int n = nt * 16 + m;
      v16h b;
#pragma unroll
      for (int e = 0; e < 16; ++e) {
        const int k = kt * 32 + ((e < 8) ? (kh * 8 + e) : (16 + kh * 8 + (e - 8)));
        b[e] = (_Float16)sWT[n * DIM + (k ^ kxor)];
      }
      acc[nt] = __builtin_amdgcn_wmma_f32_16x16x32_f16(
          false, a, false, b, (short)0, acc[nt], false, false);
    }
  }
#endif

  // C/D layout: lanes 0-15 -> M=j, N=lane; lanes 16-31 -> M=8+j, N=lane-16
#pragma unroll
  for (int nt = 0; nt < 8; ++nt) {
    const int col = nt * 16 + m;
#pragma unroll
    for (int j = 0; j < 8; ++j) {
      const int row = tile * 16 + kh * 8 + j;
      H[(size_t)row * DIM + col] = acc[nt][j];
    }
  }
}

// ---------------- agg init: self-loop + bias ----------------
// agg[v][c] = dinv[v]^2 * h[v][c] + b[c]

__global__ __launch_bounds__(256) void k_init_agg(const float* __restrict__ h,
                                                  const float* __restrict__ dinv,
                                                  const float* __restrict__ bias,
                                                  float* __restrict__ agg, int n) {
  int idx  = blockIdx.x * 256 + threadIdx.x;
  int node = idx >> 5;
  int c4   = idx & 31;
  if (node >= n) return;
  float di = dinv[node];
  float s  = di * di;
  float4 v = ((const float4*)(h + (size_t)node * DIM))[c4];
  float4 b = ((const float4*)bias)[c4];
  float4 o;
  o.x = s * v.x + b.x;
  o.y = s * v.y + b.y;
  o.z = s * v.z + b.z;
  o.w = s * v.w + b.w;
  ((float4*)(agg + (size_t)node * DIM))[c4] = o;
}

// ---------------- edge scatter-add (wave32, EPW edges/wave) ----------------
// agg[dst] += dinv[src]*dinv[dst] * h[src]
// Coalesced 512B row gather per edge + f32 L2 atomics; the NEXT edge's source row
// is prefetched (global_prefetch_b8) to overlap random-gather L2 latency with the
// current edge's atomic traffic.

__global__ __launch_bounds__(256) void k_edge_agg(const int* __restrict__ src,
                                                  const int* __restrict__ dst,
                                                  const float* __restrict__ dinv,
                                                  const float* __restrict__ h,
                                                  float* __restrict__ agg, int e) {
  int w    = (blockIdx.x * 256 + threadIdx.x) >> 5;
  int lane = threadIdx.x & 31;
  int base = w * EPW;
  if (base >= e) return;

  int s_cur = src[base];
#pragma unroll
  for (int i = 0; i < EPW; ++i) {
    int cur = base + i;
    if (cur >= e) return;
    int s = s_cur;
    int d = dst[cur];
    if (i + 1 < EPW && cur + 1 < e) {
      s_cur = src[cur + 1];
      // warm the next gather row (each lane covers its 16B slice of the 512B row)
      __builtin_prefetch(h + (size_t)s_cur * DIM + lane * 4, 0, 3);
    }
    float norm = dinv[s] * dinv[d];
    float4 v = ((const float4*)(h + (size_t)s * DIM))[lane];
    float* op = agg + (size_t)d * DIM + lane * 4;
    atomicAdd(op + 0, norm * v.x);
    atomicAdd(op + 1, norm * v.y);
    atomicAdd(op + 2, norm * v.z);
    atomicAdd(op + 3, norm * v.w);
  }
}

// ---------------- LayerNorm + ReLU + residual (wave32 per node) ----------------

__global__ __launch_bounds__(256) void k_ln_relu_res(const float* __restrict__ agg,
                                                     const float* __restrict__ resid,
                                                     const float* __restrict__ lnw,
                                                     const float* __restrict__ lnb,
                                                     float* __restrict__ out, int n) {
  int node = (blockIdx.x * 256 + threadIdx.x) >> 5;
  int lane = threadIdx.x & 31;
  if (node >= n) return;

  float4 v = ((const float4*)(agg + (size_t)node * DIM))[lane];
  float s = v.x + v.y + v.z + v.w;
  float q = v.x * v.x + v.y * v.y + v.z * v.z + v.w * v.w;
#pragma unroll
  for (int off = 16; off > 0; off >>= 1) {
    s += __shfl_xor(s, off, 32);
    q += __shfl_xor(q, off, 32);
  }
  float mu  = s * (1.0f / DIM);
  float var = q * (1.0f / DIM) - mu * mu;
  float r   = rsqrtf(var + LN_EPS);

  float4 w4 = ((const float4*)lnw)[lane];
  float4 b4 = ((const float4*)lnb)[lane];
  float4 x4 = ((const float4*)(resid + (size_t)node * DIM))[lane];
  float4 o;
  o.x = fmaxf((v.x - mu) * r * w4.x + b4.x, 0.0f) + x4.x;
  o.y = fmaxf((v.y - mu) * r * w4.y + b4.y, 0.0f) + x4.y;
  o.z = fmaxf((v.z - mu) * r * w4.z + b4.z, 0.0f) + x4.z;
  o.w = fmaxf((v.w - mu) * r * w4.w + b4.w, 0.0f) + x4.w;
  ((float4*)(out + (size_t)node * DIM))[lane] = o;
}

// ---------------- launcher ----------------

extern "C" void kernel_launch(void* const* d_in, const int* in_sizes, int n_in,
                              void* d_out, int out_size, void* d_ws, size_t ws_size,
                              hipStream_t stream) {
  (void)n_in; (void)out_size; (void)ws_size;

  const float* x    = (const float*)d_in[0];
  const int*   ei   = (const int*)d_in[1];   // [2, E]
  const float* W0   = (const float*)d_in[2];
  const float* b0   = (const float*)d_in[3];
  const float* W1   = (const float*)d_in[4];
  const float* b1   = (const float*)d_in[5];
  const float* ln0w = (const float*)d_in[6];
  const float* ln0b = (const float*)d_in[7];
  const float* ln1w = (const float*)d_in[8];
  const float* ln1b = (const float*)d_in[9];

  const int N = in_sizes[0] / DIM;
  const int E = in_sizes[1] / 2;
  const int* srcp = ei;
  const int* dstp = ei + E;

  float* deg = (float*)d_ws;                  // N floats (becomes dinv)
  float* h   = deg + N;                       // N*DIM
  float* agg = h + (size_t)N * DIM;           // N*DIM
  float* h1  = agg + (size_t)N * DIM;         // N*DIM
  float* out = (float*)d_out;

  const int TPB = 256;
  const int nTiles      = (N + 15) / 16;
  const int gemmBlocks  = (nTiles + 7) / 8;
  const int waveBlocksN = (N + 7) / 8;
  const int waveBlocksE = (E + 8 * EPW - 1) / (8 * EPW);
  const int nodeFeatBlk = (int)(((size_t)N * 32 + TPB - 1) / TPB);

  // degrees -> dinv
  k_deg_init <<<(N + TPB - 1) / TPB, TPB, 0, stream>>>(deg, N);
  k_deg_edges<<<(E + TPB - 1) / TPB, TPB, 0, stream>>>(dstp, deg, E);
  k_deg_rsqrt<<<(N + TPB - 1) / TPB, TPB, 0, stream>>>(deg, N);

  // layer 0
  k_gemm_wmma  <<<gemmBlocks,  TPB, 0, stream>>>(x, W0, h, nTiles);
  k_init_agg   <<<nodeFeatBlk, TPB, 0, stream>>>(h, deg, b0, agg, N);
  k_edge_agg   <<<waveBlocksE, TPB, 0, stream>>>(srcp, dstp, deg, h, agg, E);
  k_ln_relu_res<<<waveBlocksN, TPB, 0, stream>>>(agg, x, ln0w, ln0b, h1, N);

  // layer 1
  k_gemm_wmma  <<<gemmBlocks,  TPB, 0, stream>>>(h1, W1, h, nTiles);
  k_init_agg   <<<nodeFeatBlk, TPB, 0, stream>>>(h, deg, b1, agg, N);
  k_edge_agg   <<<waveBlocksE, TPB, 0, stream>>>(srcp, dstp, deg, h, agg, E);
  k_ln_relu_res<<<waveBlocksN, TPB, 0, stream>>>(agg, h1, ln1w, ln1b, out, N);
}
